// MoEBridge_3530463117433
// MI455X (gfx1250) — matrix operand order repaired and checked by reading, not compiled, not run on hardware
//
#include <hip/hip_runtime.h>
#include <math.h>

// ---------------- types ----------------
typedef __bf16 bf16;
typedef __attribute__((ext_vector_type(16))) __bf16 v16bf;
typedef __attribute__((ext_vector_type(8)))  float   v8f;

// ---------------- problem dims ----------------
#define B_     64
#define S_     1024
#define SRC_   4096
#define D_     1024
#define K_     64
#define H_     4096
#define E_     8
#define HEADS_ 8
#define DH_    128
#define NTOK   (B_*K_)     // 4096 latent tokens
#define NCTX   (B_*S_)     // 65536 context tokens
#define NSOFT  (NTOK*D_)   // 4194304 output elements

// ---------------- GEMM tiling ----------------
#define BM  256
#define BN  128
#define BK  32
#define BKP 40   // padded LDS row (elements) -> conflict-free b128 frag loads

// =====================================================================
// bf16 WMMA GEMM:  Out[M,N] = epilogue( A[M,K] @ Bw[K,N] )
//   A fp32 (converted while staging) or bf16 (async-DMA'd raw into LDS)
//   Bw fp32 weights, converted+transposed while staging
//   epilogue: +bias[n], exact GELU, *(*scale_ptr), *rowscale[m], +addend,
//             fp32 or bf16 store
// Requires: M%256==0, N%128==0, K%32==0 (true for all uses here)
// =====================================================================
__global__ __launch_bounds__(256)
void gemm_wmma(const void* __restrict__ Av, int a_bf16, int lda,
               const float* __restrict__ Bw, int ldb,
               const float* __restrict__ bias,
               const float* __restrict__ scale_ptr,
               const float* __restrict__ rowscale, int rs_stride,
               const float* __restrict__ addend, int lad,
               void* __restrict__ Outv, int out_bf16, int ldo,
               int M, int N, int Kd, int act)
{
  __shared__ bf16 As[2][BM][BKP];   // 40 KB
  __shared__ bf16 Bs[2][BN][BKP];   // 20 KB

  const int tid = threadIdx.x;
  const int bm0 = blockIdx.y * BM;
  const int bn0 = blockIdx.x * BN;
  const int nk  = Kd / BK;

  const float* Af = (const float*)Av;
  const bf16*  Ab = (const bf16*)Av;

  // --- A staging: bf16 -> async DMA to LDS (ASYNCcnt); fp32 -> convert ---
#define STAGE_A(bufi, kt) do {                                                \
    const int k0_ = (kt) * BK;                                                \
    if (a_bf16) {                                                             \
      _Pragma("unroll")                                                       \
      for (int p_ = 0; p_ < 4; ++p_) {                                        \
        const int r_ = p_*64 + (tid >> 2);                                    \
        const int c_ = (tid & 3) * 8;                                         \
        unsigned lds_ = (unsigned)(size_t)&As[bufi][r_][c_];                  \
        unsigned long long g_ =                                               \
            (unsigned long long)(size_t)(Ab + (size_t)(bm0 + r_) * lda + k0_ + c_); \
        asm volatile("global_load_async_to_lds_b128 %0, %1, off"              \
                     :: "v"(lds_), "v"(g_) : "memory");                       \
      }                                                                       \
    } else {                                                                  \
      _Pragma("unroll")                                                       \
      for (int p_ = 0; p_ < 8; ++p_) {                                        \
        const int r_ = p_*32 + (tid >> 3);                                    \
        const int c_ = (tid & 7) * 4;                                         \
        const float4 f_ = *(const float4*)(Af + (size_t)(bm0 + r_) * lda + k0_ + c_); \
        As[bufi][r_][c_+0] = (bf16)f_.x; As[bufi][r_][c_+1] = (bf16)f_.y;     \
        As[bufi][r_][c_+2] = (bf16)f_.z; As[bufi][r_][c_+3] = (bf16)f_.w;     \
      }                                                                       \
    }                                                                         \
  } while (0)

  // --- B staging: fp32 weights, convert + transpose to [n][k] ---
#define STAGE_B(bufi, kt) do {                                                \
    const int k0_ = (kt) * BK;                                                \
    _Pragma("unroll")                                                         \
    for (int p_ = 0; p_ < 4; ++p_) {                                          \
      const int kr_ = p_*8 + (tid >> 5);                                      \
      const int nc_ = (tid & 31) * 4;                                         \
      const float4 f_ = *(const float4*)(Bw + (size_t)(k0_ + kr_) * ldb + bn0 + nc_); \
      Bs[bufi][nc_+0][kr_] = (bf16)f_.x; Bs[bufi][nc_+1][kr_] = (bf16)f_.y;   \
      Bs[bufi][nc_+2][kr_] = (bf16)f_.z; Bs[bufi][nc_+3][kr_] = (bf16)f_.w;   \
    }                                                                         \
  } while (0)

  const int lane  = tid & 31;
  const int wv    = tid >> 5;
  const int wm    = wv & 3;        // 4 row-groups of 64 rows
  const int wn    = wv >> 2;       // 2 col-groups of 64 cols
  const int mrow0 = wm * 64;
  const int ncol0 = wn * 64;
  const int lhalf = lane >> 4;     // 0 | 1
  const int l16   = lane & 15;
  const int khalf = lhalf * 8;     // A-frag K split {0-7,16-23} / {8-15,24-31}

  v8f acc[4][4];
#pragma unroll
  for (int i = 0; i < 4; ++i)
#pragma unroll
    for (int j = 0; j < 4; ++j) acc[i][j] = (v8f){0,0,0,0,0,0,0,0};

  STAGE_A(0, 0);
  STAGE_B(0, 0);
  int buf = 0;
  for (int kt = 0; kt < nk; ++kt) {
    if (a_bf16) asm volatile("s_wait_asynccnt 0x0" ::: "memory");
    __syncthreads();
    if (kt + 1 < nk) {
      if (kt + 2 < nk)   // gfx1250 global_prefetch_b8 for weight tile after next
        __builtin_prefetch(Bw + (size_t)((kt+2)*BK + (tid >> 5)) * ldb + bn0, 0, 1);
      STAGE_A((buf ^ 1), kt + 1);
      STAGE_B((buf ^ 1), kt + 1);
    }

    v16bf afr[4], bfr[4];
#pragma unroll
    for (int i = 0; i < 4; ++i) {
      const bf16* p = &As[buf][mrow0 + i*16 + l16][0];
      union { uint4 q[2]; v16bf v; } u;
      u.q[0] = *(const uint4*)(p + khalf);        // K = khalf .. khalf+7
      u.q[1] = *(const uint4*)(p + khalf + 16);   // K = khalf+16 .. khalf+23
      afr[i] = u.v;
    }
#pragma unroll
    for (int j = 0; j < 4; ++j) {
      const bf16* p = &Bs[buf][ncol0 + j*16 + l16][0];
      union { uint4 q[2]; v16bf v; } u;
      u.q[0] = *(const uint4*)(p + khalf);          // K = khalf .. khalf+7
      u.q[1] = *(const uint4*)(p + khalf + 16);     // K = khalf+16 .. khalf+23
      bfr[j] = u.v;
    }
#pragma unroll
    for (int i = 0; i < 4; ++i)
#pragma unroll
      for (int j = 0; j < 4; ++j)
        acc[i][j] = __builtin_amdgcn_wmma_f32_16x16x32_bf16(
            false, afr[i], false, bfr[j], (short)0, acc[i][j], false, false);
    buf ^= 1;
  }
  __syncthreads();

  const float sc = scale_ptr ? *scale_ptr : 1.0f;
  float* Outf = (float*)Outv;
  bf16*  Outb = (bf16*)Outv;
#pragma unroll
  for (int i = 0; i < 4; ++i) {
#pragma unroll
    for (int j = 0; j < 4; ++j) {
      const int gn = bn0 + ncol0 + j*16 + l16;
      const float bv = bias ? bias[gn] : 0.0f;
#pragma unroll
      for (int r = 0; r < 8; ++r) {
        const int gm = bm0 + mrow0 + i*16 + lhalf*8 + r;  // C/D: M=r(+8), N=lane%16
        float x = acc[i][j][r] + bv;
        if (act) x = 0.5f * x * (1.0f + erff(x * 0.70710678118654752f));
        x *= sc;
        if (rowscale) x *= rowscale[(size_t)gm * rs_stride];
        if (addend)   x += addend[(size_t)gm * lad + gn];
        if (out_bf16) Outb[(size_t)gm * ldo + gn] = (bf16)x;
        else          Outf[(size_t)gm * ldo + gn] = x;
      }
    }
  }
#undef STAGE_A
#undef STAGE_B
}

// =====================================================================
// LayerNorm: one row per block, D_ = 1024
// =====================================================================
__global__ __launch_bounds__(256)
void layernorm_k(const float* __restrict__ x, const float* __restrict__ g,
                 const float* __restrict__ b, void* __restrict__ outv,
                 int out_bf16)
{
  __shared__ float red[256];
  const int row = blockIdx.x;
  const float* xr = x + (size_t)row * D_;
  float s = 0.f, s2 = 0.f;
  for (int d = threadIdx.x; d < D_; d += 256) { float v = xr[d]; s += v; s2 += v*v; }
  red[threadIdx.x] = s; __syncthreads();
  for (int o = 128; o > 0; o >>= 1) { if (threadIdx.x < o) red[threadIdx.x] += red[threadIdx.x+o]; __syncthreads(); }
  const float mean = red[0] * (1.0f / D_);
  __syncthreads();
  red[threadIdx.x] = s2; __syncthreads();
  for (int o = 128; o > 0; o >>= 1) { if (threadIdx.x < o) red[threadIdx.x] += red[threadIdx.x+o]; __syncthreads(); }
  const float var = red[0] * (1.0f / D_) - mean * mean;
  const float inv = rsqrtf(var + 1e-5f);
  for (int d = threadIdx.x; d < D_; d += 256) {
    float y = (xr[d] - mean) * inv * g[d] + b[d];
    if (out_bf16) ((bf16*)outv)[(size_t)row * D_ + d] = (bf16)y;
    else          ((float*)outv)[(size_t)row * D_ + d] = y;
  }
}

// =====================================================================
// Cross-attention: block = (b, head, half of 32 latents); 144KB dynamic LDS
// q fp32 [B,K,D] (pre-projected), k/v bf16 [B,S,D]; out bf16 [B,K,D]
// =====================================================================
#define KQB 32
__global__ __launch_bounds__(256)
void attention_k(const float* __restrict__ q, const bf16* __restrict__ kb,
                 const bf16* __restrict__ vb, bf16* __restrict__ o)
{
  extern __shared__ char smem_att[];
  float* qs = (float*)smem_att;       // [KQB][DH_]
  float* ps = qs + KQB * DH_;         // [KQB][S_]
  const int bid  = blockIdx.x;
  const int half = bid & 1;
  const int h    = (bid >> 1) & (HEADS_ - 1);
  const int b    = bid >> 4;
  const int kq0  = half * KQB;
  const float scale = 0.08838834764831845f;  // dh^-0.5, dh=128

  for (int i = threadIdx.x; i < KQB * DH_; i += 256) {
    const int kq = i >> 7, d = i & (DH_ - 1);
    qs[i] = q[(size_t)(b*K_ + kq0 + kq) * D_ + h*DH_ + d] * scale;
  }
  __syncthreads();

  for (int i = threadIdx.x; i < KQB * S_; i += 256) {
    const int kq = i >> 10, s = i & (S_ - 1);
    const bf16* kr = kb + (size_t)(b*S_ + s) * D_ + h*DH_;
    const float* qr = qs + kq * DH_;
    float acc = 0.f;
    for (int c = 0; c < DH_; c += 8) {
      uint4 pk = *(const uint4*)(kr + c);
      const bf16* pp = (const bf16*)&pk;
#pragma unroll
      for (int t = 0; t < 8; ++t) acc += qr[c + t] * (float)pp[t];
    }
    ps[i] = acc;
  }
  __syncthreads();

  if (threadIdx.x < KQB) {          // softmax over s per latent row
    float* pr = ps + threadIdx.x * S_;
    float mx = -1e30f;
    for (int s = 0; s < S_; ++s) mx = fmaxf(mx, pr[s]);
    float sum = 0.f;
    for (int s = 0; s < S_; ++s) { float e = expf(pr[s] - mx); pr[s] = e; sum += e; }
    const float r = 1.0f / sum;
    for (int s = 0; s < S_; ++s) pr[s] *= r;
  }
  __syncthreads();

  for (int i = threadIdx.x; i < KQB * DH_; i += 256) {
    const int kq = i >> 7, d = i & (DH_ - 1);
    const float* pr = ps + kq * S_;
    const bf16* vc = vb + (size_t)b * S_ * D_ + h*DH_ + d;
    float acc = 0.f;
    for (int s = 0; s < S_; ++s) acc += pr[s] * (float)vc[(size_t)s * D_];
    o[(size_t)(b*K_ + kq0 + kq) * D_ + h*DH_ + d] = (bf16)acc;
  }
}

// =====================================================================
// Router: logits -> softmax(E=8) -> top-2 renormalized weights
// =====================================================================
__global__ __launch_bounds__(256)
void router_k(const float* __restrict__ lat, const float* __restrict__ rw,
              float* __restrict__ probs, float* __restrict__ wmap,
              int* __restrict__ sel)
{
  __shared__ float red[256 * E_];
  const int row = blockIdx.x;
  const float* xr = lat + (size_t)row * D_;
  float p[E_];
#pragma unroll
  for (int e = 0; e < E_; ++e) p[e] = 0.f;
  for (int d = threadIdx.x; d < D_; d += 256) {
    const float v = xr[d];
#pragma unroll
    for (int e = 0; e < E_; ++e) p[e] += v * rw[d*E_ + e];
  }
#pragma unroll
  for (int e = 0; e < E_; ++e) red[threadIdx.x*E_ + e] = p[e];
  __syncthreads();
  for (int o = 128; o > 0; o >>= 1) {
    if (threadIdx.x < o)
#pragma unroll
      for (int e = 0; e < E_; ++e) red[threadIdx.x*E_ + e] += red[(threadIdx.x+o)*E_ + e];
    __syncthreads();
  }
  if (threadIdx.x == 0) {
    float lg[E_]; float mx = -1e30f;
#pragma unroll
    for (int e = 0; e < E_; ++e) { lg[e] = red[e]; mx = fmaxf(mx, lg[e]); }
    float sum = 0.f;
#pragma unroll
    for (int e = 0; e < E_; ++e) { lg[e] = expf(lg[e] - mx); sum += lg[e]; }
    const float rs = 1.0f / sum;
#pragma unroll
    for (int e = 0; e < E_; ++e) { lg[e] *= rs; probs[row*E_ + e] = lg[e]; }
    int i0 = 0;
#pragma unroll
    for (int e = 1; e < E_; ++e) if (lg[e] > lg[i0]) i0 = e;
    int i1 = (i0 == 0) ? 1 : 0;
#pragma unroll
    for (int e = 0; e < E_; ++e) if (e != i0 && lg[e] > lg[i1]) i1 = e;
    const float tsum = lg[i0] + lg[i1] + 1e-8f;
#pragma unroll
    for (int e = 0; e < E_; ++e) wmap[row*E_ + e] = 0.f;
    wmap[row*E_ + i0] = lg[i0] / tsum;
    wmap[row*E_ + i1] = lg[i1] / tsum;
    sel[row*2] = i0; sel[row*2 + 1] = i1;
  }
}

// =====================================================================
// small utility kernels
// =====================================================================
__global__ void zero_f32(float* __restrict__ p, int n) {
  int i = blockIdx.x * blockDim.x + threadIdx.x;
  if (i < n) p[i] = 0.f;
}
__global__ void bcast_latents(const float* __restrict__ lat0, float* __restrict__ lat) {
  int i = blockIdx.x * blockDim.x + threadIdx.x;
  if (i < NSOFT) lat[i] = lat0[i % (K_*D_)];
}
__global__ void tobf16_k(const float* __restrict__ in, bf16* __restrict__ out, int n) {
  int i = blockIdx.x * blockDim.x + threadIdx.x;
  if (i < n) out[i] = (bf16)in[i];
}

// soft = (lat+exp_out)/rms * output_scale, written straight to d_out
__global__ __launch_bounds__(256)
void finalize_k(const float* __restrict__ lat, const float* __restrict__ expo,
                const float* __restrict__ oscale, float* __restrict__ out)
{
  __shared__ float red[256];
  const int row = blockIdx.x;
  const size_t base = (size_t)row * D_;
  float v[4]; float s2 = 0.f;
#pragma unroll
  for (int t = 0; t < 4; ++t) {
    const int d = threadIdx.x + t*256;
    const float x = lat[base + d] + expo[base + d];
    v[t] = x; s2 += x*x;
  }
  red[threadIdx.x] = s2; __syncthreads();
  for (int o = 128; o > 0; o >>= 1) { if (threadIdx.x < o) red[threadIdx.x] += red[threadIdx.x+o]; __syncthreads(); }
  const float rms = sqrtf(red[0] * (1.0f / D_) + 1e-8f);
  const float sc  = oscale[0] / rms;
#pragma unroll
  for (int t = 0; t < 4; ++t) {
    const int d = threadIdx.x + t*256;
    out[base + d] = v[t] * sc;
  }
}

// aux_loss, entropy, z_var -- single block, deterministic, double precision
__global__ __launch_bounds__(256)
void stats_k(const float* __restrict__ probs, const int* __restrict__ sel,
             const float* __restrict__ soft, float* __restrict__ out3)
{
  __shared__ double dred[256];
  __shared__ double fP[E_], fC[E_], misc[3];
  const int tid = threadIdx.x;
  double pP[E_], pC[E_];
#pragma unroll
  for (int e = 0; e < E_; ++e) { pP[e] = 0.0; pC[e] = 0.0; }
  double ent = 0.0;
  for (int row = tid; row < NTOK; row += 256) {
#pragma unroll
    for (int e = 0; e < E_; ++e) {
      const float p = probs[row*E_ + e];
      pP[e] += (double)p;
      ent   -= (double)(p * logf(p + 1e-8f));
    }
    pC[sel[row*2]]     += 1.0;
    pC[sel[row*2 + 1]] += 1.0;
  }
  for (int e = 0; e < E_; ++e) {
    dred[tid] = pP[e]; __syncthreads();
    for (int o = 128; o > 0; o >>= 1) { if (tid < o) dred[tid] += dred[tid+o]; __syncthreads(); }
    if (tid == 0) fP[e] = dred[0];
    __syncthreads();
    dred[tid] = pC[e]; __syncthreads();
    for (int o = 128; o > 0; o >>= 1) { if (tid < o) dred[tid] += dred[tid+o]; __syncthreads(); }
    if (tid == 0) fC[e] = dred[0];
    __syncthreads();
  }
  dred[tid] = ent; __syncthreads();
  for (int o = 128; o > 0; o >>= 1) { if (tid < o) dred[tid] += dred[tid+o]; __syncthreads(); }
  if (tid == 0) misc[0] = dred[0];
  __syncthreads();

  double s = 0.0, s2 = 0.0;
  for (size_t i = tid; i < (size_t)NSOFT; i += 256) {
    const double x = (double)soft[i];
    s += x; s2 += x*x;
  }
  dred[tid] = s; __syncthreads();
  for (int o = 128; o > 0; o >>= 1) { if (tid < o) dred[tid] += dred[tid+o]; __syncthreads(); }
  if (tid == 0) misc[1] = dred[0];
  __syncthreads();
  dred[tid] = s2; __syncthreads();
  for (int o = 128; o > 0; o >>= 1) { if (tid < o) dred[tid] += dred[tid+o]; __syncthreads(); }
  if (tid == 0) misc[2] = dred[0];
  __syncthreads();

  if (tid == 0) {
    double aux = 0.0;
#pragma unroll
    for (int e = 0; e < E_; ++e) aux += (fC[e] / NTOK) * (fP[e] / NTOK);
    out3[0] = (float)(0.01 * E_ * aux);
    out3[1] = (float)(misc[0] / NTOK);
    const double n = (double)NSOFT;
    const double mean = misc[1] / n;
    out3[2] = (float)((misc[2] - n * mean * mean) / (n - 1.0));
  }
}

// =====================================================================
// host-side orchestration
// =====================================================================
static inline void gemm(hipStream_t st, const void* A, int a_bf16, int lda,
                        const float* Bw, int ldb, const float* bias,
                        const float* scale_ptr, const float* rowscale, int rs_stride,
                        const float* addend, int lad,
                        void* Out, int out_bf16, int ldo,
                        int M, int N, int Kd, int act)
{
  dim3 g(N / BN, M / BM), bl(256);
  gemm_wmma<<<g, bl, 0, st>>>(A, a_bf16, lda, Bw, ldb, bias, scale_ptr,
                              rowscale, rs_stride, addend, lad,
                              Out, out_bf16, ldo, M, N, Kd, act);
}

extern "C" void kernel_launch(void* const* d_in, const int* in_sizes, int n_in,
                              void* d_out, int out_size, void* d_ws, size_t ws_size,
                              hipStream_t stream) {
  (void)in_sizes; (void)n_in; (void)out_size; (void)ws_size;
  // ---- inputs (setup_inputs dict order, params flattened) ----
  const float* src      = (const float*)d_in[0];    // [B,S,SRC]
  const float* in_w     = (const float*)d_in[1];    // [SRC,D]
  const float* in_b     = (const float*)d_in[2];    // [D]
  const float* latents  = (const float*)d_in[3];    // [K,D]
  // layers: 14 tensors each, starting at 4
  const float* router_w = (const float*)d_in[32];   // [D,E]
  const float* exp_w1   = (const float*)d_in[33];   // [E,D,H]
  const float* exp_b1   = (const float*)d_in[34];   // [E,H]
  const float* exp_w2   = (const float*)d_in[35];   // [E,H,D]
  const float* exp_b2   = (const float*)d_in[36];   // [E,D]
  const float* sh_w1    = (const float*)d_in[37];   // [D,2D]
  const float* sh_b1    = (const float*)d_in[38];   // [2D]
  const float* sh_w2    = (const float*)d_in[39];   // [2D,D]
  const float* sh_b2    = (const float*)d_in[40];   // [D]
  const float* sh_wt    = (const float*)d_in[41];   // scalar
  const float* out_sc   = (const float*)d_in[42];   // scalar

  // ---- workspace layout (bytes) ----
  char* ws = (char*)d_ws;
  float* ctx  = (float*)(ws + 0);                         // NCTX*D f32  (256MB)
  bf16*  cin  = (bf16*) (ws + 268435456ull);              // NCTX*D bf16 (128MB)
  bf16*  kbuf = (bf16*) (ws + 402653184ull);              // NCTX*D bf16
  bf16*  vbuf = (bf16*) (ws + 536870912ull);              // NCTX*D bf16
  float* lat  = (float*)(ws + 671088640ull);              // NTOK*D f32
  bf16*  lnb  = (bf16*) (ws + 687865856ull);              // NTOK*D bf16 (LN outs)
  float* qo   = (float*)(ws + 704643072ull);              // NTOK*D f32
  bf16*  hid  = (bf16*) (ws + 721420288ull);              // NTOK*H bf16 (32MB)
  float* expo = (float*)(ws + 754974720ull);              // NTOK*D f32
  float* probs= (float*)(ws + 771751936ull);              // NTOK*E f32
  float* wmap = (float*)(ws + 771883008ull);              // NTOK*E f32
  int*   sel  = (int*)  (ws + 772014080ull);              // NTOK*2 i32
  bf16*  latb = (bf16*) (ws + 772046848ull);              // NTOK*D bf16
  bf16*  aob  = (bf16*) (ws + 780435456ull);              // NTOK*D bf16 (attn out)

  // 1) lat = broadcast(latents)
  bcast_latents<<<NSOFT/256, 256, 0, stream>>>(latents, lat);

  // 2) ctx = src @ in_w + in_b     (550 GFLOP, bf16 WMMA, fp32-A path)
  gemm(stream, src, 0, SRC_, in_w, D_, in_b, nullptr, nullptr, 0,
       nullptr, 0, ctx, 0, D_, NCTX, D_, SRC_, 0);

  // 3) perceiver layers
  for (int l = 0; l < 2; ++l) {
    const float** lp = (const float**)(d_in + 4 + 14*l);
    const float *ln1_g = lp[0], *ln1_b = lp[1], *lnc_g = lp[2], *lnc_b = lp[3];
    const float *wq = lp[4], *wk = lp[5], *wv = lp[6], *wo = lp[7];
    const float *ln2_g = lp[8], *ln2_b = lp[9];
    const float *w1 = lp[10], *b1 = lp[11], *w2 = lp[12], *b2 = lp[13];

    layernorm_k<<<NTOK, 256, 0, stream>>>(lat, ln1_g, ln1_b, lnb, 1);       // q_in bf16
    layernorm_k<<<NCTX, 256, 0, stream>>>(ctx, lnc_g, lnc_b, cin, 1);       // c_in bf16

    gemm(stream, lnb, 1, D_, wq, D_, nullptr, nullptr, nullptr, 0,
         nullptr, 0, qo, 0, D_, NTOK, D_, D_, 0);                           // q (f32)
    gemm(stream, cin, 1, D_, wk, D_, nullptr, nullptr, nullptr, 0,
         nullptr, 0, kbuf, 1, D_, NCTX, D_, D_, 0);                         // k (bf16)
    gemm(stream, cin, 1, D_, wv, D_, nullptr, nullptr, nullptr, 0,
         nullptr, 0, vbuf, 1, D_, NCTX, D_, D_, 0);                         // v (bf16)

    attention_k<<<B_*HEADS_*2, 256, (KQB*DH_ + KQB*S_)*sizeof(float), stream>>>(
        qo, kbuf, vbuf, aob);                                               // o -> bf16

    gemm(stream, aob, 1, D_, wo, D_, nullptr, nullptr, nullptr, 0,
         lat, D_, lat, 0, D_, NTOK, D_, D_, 0);                             // lat += o@wo

    layernorm_k<<<NTOK, 256, 0, stream>>>(lat, ln2_g, ln2_b, lnb, 1);
    gemm(stream, lnb, 1, D_, w1, H_, b1, nullptr, nullptr, 0,
         nullptr, 0, hid, 1, H_, NTOK, H_, D_, 1);                          // gelu(x@w1+b1)
    gemm(stream, hid, 1, H_, w2, D_, b2, nullptr, nullptr, 0,
         lat, D_, lat, 0, D_, NTOK, D_, H_, 0);                             // lat += h@w2+b2
  }

  // 4) router
  router_k<<<NTOK, 256, 0, stream>>>(lat, router_w, probs, wmap, sel);

  // 5) experts (dense over all tokens, combined with wmap in GEMM epilogue)
  zero_f32<<<NSOFT/256, 256, 0, stream>>>(expo, NSOFT);
  tobf16_k<<<NSOFT/256, 256, 0, stream>>>(lat, latb, NSOFT);
  for (int e = 0; e < E_; ++e) {
    gemm(stream, latb, 1, D_, exp_w1 + (size_t)e*D_*H_, H_, exp_b1 + (size_t)e*H_,
         nullptr, nullptr, 0, nullptr, 0, hid, 1, H_, NTOK, H_, D_, 1);
    gemm(stream, hid, 1, H_, exp_w2 + (size_t)e*H_*D_, D_, exp_b2 + (size_t)e*D_,
         nullptr, wmap + e, E_, expo, D_, expo, 0, D_, NTOK, D_, H_, 0);
  }
  // shared expert (hidden 2D), scaled by device scalar shared_weight
  gemm(stream, latb, 1, D_, sh_w1, 2*D_, sh_b1, nullptr, nullptr, 0,
       nullptr, 0, hid, 1, 2*D_, NTOK, 2*D_, D_, 1);
  gemm(stream, hid, 1, 2*D_, sh_w2, D_, sh_b2, sh_wt, nullptr, 0,
       expo, D_, expo, 0, D_, NTOK, D_, 2*D_, 0);

  // 6) soft = (lat + expo)/rms * output_scale -> d_out ; then scalar stats
  finalize_k<<<NTOK, 256, 0, stream>>>(lat, expo, out_sc, (float*)d_out);
  stats_k<<<1, 256, 0, stream>>>(probs, sel, (float*)d_out, (float*)d_out + NSOFT);
}
